// PointerGeneratorDecoder_61641370632429
// MI455X (gfx1250) — matrix operand order repaired and checked by reading, not compiled, not run on hardware
//
#include <hip/hip_runtime.h>
#include <hip/hip_bf16.h>
#include <cstdint>
#include <cstddef>

// ---------------- CDNA5 WMMA types ----------------
typedef __bf16 bf16_t;
typedef __attribute__((ext_vector_type(16))) __bf16 v16bf;
typedef __attribute__((ext_vector_type(2)))  __bf16 v2bf;
typedef __attribute__((ext_vector_type(8)))  float  v8f;

#define DEV static __device__ __forceinline__
#define USE_ASYNC_LDS 1

// Sizes
#define BB   64
#define SS   2048
#define HH   512
#define ENCD 1024
#define EMBD 128
#define VV   50000

// native hardware f32->bf16 (RNE) conversion
DEV bf16_t f2bf(float f) { return (bf16_t)f; }

// pack two f32 -> two bf16 in one dword (lets clang select packed cvt)
DEV uint32_t pack_bf2(float a, float b) {
    v2bf v = { (bf16_t)a, (bf16_t)b };
    union { v2bf v; uint32_t u; } c; c.v = v;
    return c.u;
}

DEV v8f v8zero() { v8f z; for (int i = 0; i < 8; ++i) z[i] = 0.f; return z; }

union FragU { v16bf v; uint4 q[2]; };

// ---- CDNA5 async global->LDS copy (16B per lane), tracked by ASYNCcnt ----
DEV void async_copy_b128(bf16_t* ldsdst, const bf16_t* gsrc) {
#if USE_ASYNC_LDS
    uint32_t loff = (uint32_t)(uintptr_t)ldsdst;   // LDS aperture: addr[31:0] = LDS offset
    asm volatile("global_load_async_to_lds_b128 %0, %1, off"
                 :: "v"(loff), "v"(gsrc) : "memory");
#else
    *(uint4*)ldsdst = *(const uint4*)gsrc;
#endif
}
DEV void wait_async() {
#if USE_ASYNC_LDS
    asm volatile("s_wait_asynccnt 0x0" ::: "memory");
#endif
}

// A fragment: 16x32 bf16 tile, LDS layout [16 rows][32 K] row-major.
DEV v16bf load_a_frag(const bf16_t* sA, int mtile, int lane) {
    int r = lane & 15, s = lane >> 4;
    const bf16_t* row = sA + (mtile * 16 + r) * 32;
    FragU f;
    f.q[0] = *(const uint4*)(row + s * 8);
    f.q[1] = *(const uint4*)(row + 16 + s * 8);
    return f.v;
}

// B fragment: 32x16 bf16 tile staged N-major in LDS: [16 N][32 K].
DEV v16bf load_b_frag(const bf16_t* sB, int ntile, int lane) {
    int c = lane & 15, kb = (lane >> 4) * 8;
    const bf16_t* row = sB + (ntile * 16 + c) * 32;
    FragU f;
    f.q[0] = *(const uint4*)(row + kb);
    f.q[1] = *(const uint4*)(row + 16 + kb);
    return f.v;
}

// vectorized f32->bf16 tile stage: rows*32 f32 (row stride lda) -> LDS [rows][32] bf16
DEV void stage_f32_tile(const float* __restrict__ src, size_t lda, int k0,
                        bf16_t* __restrict__ dst, int rows, int t, int nthreads) {
    int nvec = rows * 8;   // float4s per tile
    for (int i = t; i < nvec; i += nthreads) {
        int r = i >> 3, c4 = i & 7;
        float4 v = *(const float4*)(src + (size_t)r * lda + k0 + c4 * 4);
        uint2 p;
        p.x = pack_bf2(v.x, v.y);
        p.y = pack_bf2(v.z, v.w);
        *(uint2*)(dst + r * 32 + c4 * 4) = p;
    }
}

DEV float block_reduce(float v, float* red, bool ismax) {
    for (int m = 16; m >= 1; m >>= 1) {
        float o = __shfl_xor(v, m, 32);
        v = ismax ? fmaxf(v, o) : v + o;
    }
    int w = threadIdx.x >> 5;
    if ((threadIdx.x & 31) == 0) red[w] = v;
    __syncthreads();
    float r = red[0];
    int nw = blockDim.x >> 5;
    for (int i = 1; i < nw; ++i) r = ismax ? fmaxf(r, red[i]) : r + red[i];
    __syncthreads();
    return r;
}

DEV float sigm(float x) { return 1.f / (1.f + __expf(-x)); }

// ---------------- weight conversion: dst[n][0:Ka]=A[n], dst[n][Ka:Ka+Kb]=B[n] ----------------
__global__ void cvt_cat_bf16_kernel(const float* __restrict__ A, int Ka,
                                    const float* __restrict__ Bm, int Kb,
                                    bf16_t* __restrict__ dst, int N) {
    int K = Ka + Kb;
    long total = (long)N * K;
    for (long i = (long)blockIdx.x * blockDim.x + threadIdx.x; i < total;
         i += (long)gridDim.x * blockDim.x) {
        int n = (int)(i / K), k = (int)(i % K);
        float v = (k < Ka) ? A[(long)n * Ka + k] : Bm[(long)n * Kb + (k - Ka)];
        dst[i] = f2bf(v);
    }
}

__global__ void zero_f32_kernel(float* p, long n) {
    for (long i = (long)blockIdx.x * blockDim.x + threadIdx.x; i < n;
         i += (long)gridDim.x * blockDim.x) p[i] = 0.f;
}

// ---------------- generic 64-row WMMA GEMM: C[64xN] = A[64xK](f32) * B^T, B is [N][K] ----------------
template <bool B_IS_F32>
__global__ __launch_bounds__(256) void gemm64_kernel(
    const float* __restrict__ A, int lda,
    const void* __restrict__ Bv, int ldb,
    const float* __restrict__ bias0, const float* __restrict__ bias1,
    float* __restrict__ C, int ldc, int N, int K) {
    __shared__ __align__(16) bf16_t sA[64 * 32];
    __shared__ __align__(16) bf16_t sB[256 * 32];
    int t = threadIdx.x, lane = t & 31, wave = t >> 5;
    int wm = wave >> 2, wn = wave & 3;
    int n0 = blockIdx.x * 256;

    v8f acc[2][4];
    for (int mi = 0; mi < 2; ++mi)
        for (int ni = 0; ni < 4; ++ni) acc[mi][ni] = v8zero();

    for (int k0 = 0; k0 < K; k0 += 32) {
        stage_f32_tile(A, (size_t)lda, k0, sA, 64, t, 256);
        {
            int gn = n0 + t; if (gn >= N) gn = 0;
            if constexpr (B_IS_F32) {
                const float* src = (const float*)Bv + (size_t)gn * ldb + k0;
                uint2* d = (uint2*)(sB + t * 32);
                #pragma unroll
                for (int c4 = 0; c4 < 8; ++c4) {
                    float4 v = *(const float4*)(src + c4 * 4);
                    uint2 p;
                    p.x = pack_bf2(v.x, v.y);
                    p.y = pack_bf2(v.z, v.w);
                    d[c4] = p;
                }
            } else {
                const bf16_t* src = (const bf16_t*)Bv + (size_t)gn * ldb + k0;
                bf16_t* dst = sB + t * 32;
                async_copy_b128(dst,      src);
                async_copy_b128(dst + 8,  src + 8);
                async_copy_b128(dst + 16, src + 16);
                async_copy_b128(dst + 24, src + 24);
                wait_async();
            }
        }
        __syncthreads();
        for (int mi = 0; mi < 2; ++mi) {
            v16bf a = load_a_frag(sA, wm * 2 + mi, lane);
            for (int ni = 0; ni < 4; ++ni) {
                v16bf b = load_b_frag(sB, wn * 4 + ni, lane);
                acc[mi][ni] = __builtin_amdgcn_wmma_f32_16x16x32_bf16(
                    false, a, false, b, (short)0, acc[mi][ni], false, false);
            }
        }
        __syncthreads();
    }

    int sel = lane >> 4, cl = lane & 15;
    for (int mi = 0; mi < 2; ++mi)
        for (int ni = 0; ni < 4; ++ni)
            for (int v = 0; v < 8; ++v) {
                int row = wm * 32 + mi * 16 + sel * 8 + v;
                int col = n0 + wn * 64 + ni * 16 + cl;
                if (col < N) {
                    float r = acc[mi][ni][v];
                    if (bias0) r += bias0[col];
                    if (bias1) r += bias1[col];
                    C[(size_t)row * ldc + col] = r;
                }
            }
}

// ---------------- fused attention scores: scores[bs] += sum_h tanh(enc@Wh + dec + cov*Wc)*v ----------------
__global__ __launch_bounds__(256) void attn_scores_kernel(
    const float* __restrict__ enc, const bf16_t* __restrict__ Wh,
    const float* __restrict__ dec_feat, const float* __restrict__ coverage,
    const float* __restrict__ wc, const float* __restrict__ va,
    float* __restrict__ scores) {
    __shared__ __align__(16) bf16_t sA[64 * 32];
    __shared__ __align__(16) bf16_t sB[256 * 32];
    __shared__ float sc[64];
    int t = threadIdx.x, lane = t & 31, wave = t >> 5;
    int wm = wave >> 2, wn = wave & 3;
    long row0 = (long)blockIdx.x * 64;
    int n0 = blockIdx.y * 256;
    if (t < 64) sc[t] = 0.f;

    v8f acc[2][4];
    for (int mi = 0; mi < 2; ++mi)
        for (int ni = 0; ni < 4; ++ni) acc[mi][ni] = v8zero();

    const float* Abase = enc + (size_t)row0 * ENCD;
    for (int k0 = 0; k0 < ENCD; k0 += 32) {
        stage_f32_tile(Abase, (size_t)ENCD, k0, sA, 64, t, 256);
        {
            const bf16_t* src = Wh + (size_t)(n0 + t) * ENCD + k0;
            bf16_t* dst = sB + t * 32;
            async_copy_b128(dst,      src);
            async_copy_b128(dst + 8,  src + 8);
            async_copy_b128(dst + 16, src + 16);
            async_copy_b128(dst + 24, src + 24);
            wait_async();
        }
        __syncthreads();
        for (int mi = 0; mi < 2; ++mi) {
            v16bf a = load_a_frag(sA, wm * 2 + mi, lane);
            for (int ni = 0; ni < 4; ++ni) {
                v16bf b = load_b_frag(sB, wn * 4 + ni, lane);
                acc[mi][ni] = __builtin_amdgcn_wmma_f32_16x16x32_bf16(
                    false, a, false, b, (short)0, acc[mi][ni], false, false);
            }
        }
        __syncthreads();
    }

    int sel = lane >> 4, cl = lane & 15;
    for (int mi = 0; mi < 2; ++mi) {
        for (int v = 0; v < 8; ++v) {
            int lrow = wm * 32 + mi * 16 + sel * 8 + v;
            long rowg = row0 + lrow;
            int b = (int)(rowg >> 11);             // /S
            float cov = coverage[rowg];
            float p = 0.f;
            for (int ni = 0; ni < 4; ++ni) {
                int h = n0 + wn * 64 + ni * 16 + cl;
                float e = acc[mi][ni][v] + dec_feat[b * HH + h] + cov * wc[h];
                p += tanhf(e) * va[h];
            }
            for (int m = 1; m < 16; m <<= 1) p += __shfl_xor(p, m, 32);
            if (cl == 0) atomicAdd(&sc[lrow], p);
        }
    }
    __syncthreads();
    if (t < 64) atomicAdd(&scores[row0 + t], sc[t]);
}

// ---------------- small element-wise / reduction kernels ----------------
__global__ void embed_concat_kernel(const int* __restrict__ tok, const float* __restrict__ embw,
                                    const float* __restrict__ ctxv, const float* __restrict__ h0,
                                    float* __restrict__ xcat0, float* __restrict__ out_in) {
    int b = blockIdx.x, t = threadIdx.x;
    int id = tok[b];
    if (t < EMBD) {
        float e = embw[(long)id * EMBD + t];
        xcat0[b * 1664 + t] = e;
        out_in[b * 1664 + HH + ENCD + t] = e;
    }
    for (int j = t; j < ENCD; j += 256) xcat0[b * 1664 + EMBD + j] = ctxv[b * ENCD + j];
    for (int j = t; j < HH; j += 256)   xcat0[b * 1664 + EMBD + ENCD + j] = h0[b * HH + j];
}

__global__ void lstm_act_kernel(const float* __restrict__ g, const float* __restrict__ cprev,
                                float* __restrict__ h_out, float* __restrict__ c_out,
                                float* __restrict__ h_copy, int hld) {
    int b = blockIdx.x, h = threadIdx.x;   // 512 threads
    const float* gr = g + b * 4 * HH;
    float i = sigm(gr[h]);
    float f = sigm(gr[HH + h]);
    float gg = tanhf(gr[2 * HH + h]);
    float o = sigm(gr[3 * HH + h]);
    float c2 = f * cprev[b * HH + h] + i * gg;
    float h2 = o * tanhf(c2);
    h_out[b * HH + h] = h2;
    c_out[b * HH + h] = c2;
    if (h_copy) h_copy[(size_t)b * hld + h] = h2;
}

__global__ void copy_h01_kernel(const float* __restrict__ h0, float* __restrict__ xcat1) {
    int b = blockIdx.x, h = threadIdx.x;   // 512 threads; layer-1 prev hidden
    xcat1[b * 1024 + HH + h] = h0[BB * HH + b * HH + h];
}

__global__ __launch_bounds__(256) void attn_softmax_kernel(
    const float* __restrict__ scores, const int* __restrict__ mask,
    const float* __restrict__ coverage, float* __restrict__ attn, float* __restrict__ cov_new) {
    __shared__ float red[8];
    int b = blockIdx.x, t = threadIdx.x;
    const float* srow = scores + (long)b * SS;
    float mx = -1e30f;
    for (int s = t; s < SS; s += 256) {
        float v = (mask[b * SS + s] == 0) ? -10000.f : srow[s];
        mx = fmaxf(mx, v);
    }
    mx = block_reduce(mx, red, true);
    float sum = 0.f;
    for (int s = t; s < SS; s += 256) {
        float v = (mask[b * SS + s] == 0) ? -10000.f : srow[s];
        sum += __expf(v - mx);
    }
    sum = block_reduce(sum, red, false);
    float inv = 1.f / sum;
    for (int s = t; s < SS; s += 256) {
        float v = (mask[b * SS + s] == 0) ? -10000.f : srow[s];
        float a = __expf(v - mx) * inv;
        attn[b * SS + s] = a;
        cov_new[b * SS + s] = coverage[b * SS + s] + a;
    }
}

__global__ __launch_bounds__(256) void context_kernel(
    const float* __restrict__ attn, const float* __restrict__ enc,
    float* __restrict__ o_ctx, float* __restrict__ out_in) {
    __shared__ float sa[SS];
    int b = blockIdx.x;
    int e = blockIdx.y * 256 + threadIdx.x;
    for (int s = threadIdx.x; s < SS; s += 256) sa[s] = attn[(long)b * SS + s];
    __syncthreads();
    const float* er = enc + (size_t)b * SS * ENCD + e;
    float acc = 0.f;
    #pragma unroll 8
    for (int s = 0; s < SS; ++s) acc += sa[s] * er[(size_t)s * ENCD];
    o_ctx[b * ENCD + e] = acc;
    out_in[b * 1664 + HH + e] = acc;
}

__global__ __launch_bounds__(256) void pgen_kernel(
    const float* __restrict__ out_in, const float* __restrict__ pgW,
    const float* __restrict__ pgb, float* __restrict__ o_pgen) {
    __shared__ float red[8];
    int b = blockIdx.x, t = threadIdx.x;
    float s = 0.f;
    for (int k = t; k < 1664; k += 256) s += out_in[b * 1664 + k] * pgW[k];
    s = block_reduce(s, red, false);
    if (t == 0) o_pgen[b] = sigm(s + pgb[0]);
}

__global__ __launch_bounds__(256) void vocab_softmax_kernel(
    const float* __restrict__ logits, const float* __restrict__ pgen,
    float* __restrict__ finalp) {
    __shared__ float red[8];
    int b = blockIdx.x, t = threadIdx.x;
    const float* row = logits + (size_t)b * VV;
    float mx = -1e30f;
    for (int i = t; i < VV; i += 256) mx = fmaxf(mx, row[i]);
    mx = block_reduce(mx, red, true);
    float sum = 0.f;
    for (int i = t; i < VV; i += 256) sum += __expf(row[i] - mx);
    sum = block_reduce(sum, red, false);
    float scale = pgen[b] / sum;
    for (int i = t; i < VV; i += 256)
        finalp[(size_t)b * VV + i] = __expf(row[i] - mx) * scale;
}

__global__ __launch_bounds__(256) void scatter_copy_kernel(
    const int* __restrict__ ids, const float* __restrict__ attn,
    const float* __restrict__ pgen, float* __restrict__ finalp) {
    int b = blockIdx.x;
    int s = blockIdx.y * 256 + threadIdx.x;
    float w = (1.f - pgen[b]) * attn[b * SS + s];
    int tokn = ids[b * SS + s];
    atomicAdd(&finalp[(size_t)b * VV + tokn], w);
}

// ---------------- launcher ----------------
extern "C" void kernel_launch(void* const* d_in, const int* in_sizes, int n_in,
                              void* d_out, int out_size, void* d_ws, size_t ws_size,
                              hipStream_t stream) {
    const int*   tok  = (const int*)d_in[0];
    const float* h0   = (const float*)d_in[1];
    const float* c0   = (const float*)d_in[2];
    const float* enc  = (const float*)d_in[3];
    const int*   mask = (const int*)d_in[4];
    const float* ctxv = (const float*)d_in[5];
    const float* cov  = (const float*)d_in[6];
    const int*   eids = (const int*)d_in[7];
    const float* embw = (const float*)d_in[8];
    const float* Wih0 = (const float*)d_in[9];
    const float* Whh0 = (const float*)d_in[10];
    const float* bih0 = (const float*)d_in[11];
    const float* bhh0 = (const float*)d_in[12];
    const float* Wih1 = (const float*)d_in[13];
    const float* Whh1 = (const float*)d_in[14];
    const float* bih1 = (const float*)d_in[15];
    const float* bhh1 = (const float*)d_in[16];
    const float* attWh = (const float*)d_in[17];
    const float* attWd = (const float*)d_in[18];
    const float* attWc = (const float*)d_in[19];
    const float* attv  = (const float*)d_in[20];
    const float* outW  = (const float*)d_in[21];
    const float* outb  = (const float*)d_in[22];
    const float* pgW   = (const float*)d_in[23];
    const float* pgb   = (const float*)d_in[24];

    float* out = (float*)d_out;
    float* o_final = out;                       // 64*50000
    float* o_hN   = o_final + (size_t)BB * VV;  // 2*64*512
    float* o_cN   = o_hN + 2 * BB * HH;
    float* o_ctx  = o_cN + 2 * BB * HH;         // 64*1024
    float* o_attn = o_ctx + BB * ENCD;          // 64*2048
    float* o_pgen = o_attn + BB * SS;           // 64
    float* o_cov  = o_pgen + BB;                // 64*2048

    // workspace carve
    char* wp = (char*)d_ws;
    auto alloc = [&](size_t bytes) -> void* {
        void* p = (void*)wp;
        wp += (bytes + 255) & ~(size_t)255;
        return p;
    };
    float*  xcat0  = (float*)alloc((size_t)BB * 1664 * 4);
    float*  xcat1  = (float*)alloc((size_t)BB * 1024 * 4);
    float*  g0     = (float*)alloc((size_t)BB * 2048 * 4);
    float*  g1     = (float*)alloc((size_t)BB * 2048 * 4);
    float*  decf   = (float*)alloc((size_t)BB * HH * 4);
    float*  scores = (float*)alloc((size_t)BB * SS * 4);
    float*  out_in = (float*)alloc((size_t)BB * 1664 * 4);
    float*  logits = (float*)alloc((size_t)BB * VV * 4);
    bf16_t* Wc0    = (bf16_t*)alloc((size_t)2048 * 1664 * 2);
    bf16_t* Wc1    = (bf16_t*)alloc((size_t)2048 * 1024 * 2);
    bf16_t* Whb    = (bf16_t*)alloc((size_t)HH * ENCD * 2);
    bf16_t* Wdb    = (bf16_t*)alloc((size_t)HH * HH * 2);

    // 1. weight conversions + zero scores + input concat
    cvt_cat_bf16_kernel<<<1024, 256, 0, stream>>>(Wih0, 1152, Whh0, 512, Wc0, 2048);
    cvt_cat_bf16_kernel<<<1024, 256, 0, stream>>>(Wih1, 512,  Whh1, 512, Wc1, 2048);
    cvt_cat_bf16_kernel<<<512, 256, 0, stream>>>(attWh, 1024, attWh, 0, Whb, 512);
    cvt_cat_bf16_kernel<<<256, 256, 0, stream>>>(attWd, 512,  attWd, 0, Wdb, 512);
    zero_f32_kernel<<<512, 256, 0, stream>>>(scores, (long)BB * SS);
    embed_concat_kernel<<<BB, 256, 0, stream>>>(tok, embw, ctxv, h0, xcat0, out_in);

    // 2-3. LSTM layer 0: g0 = [emb|ctx|h0_l0] @ [Wih0|Whh0]^T + b
    gemm64_kernel<false><<<8, 256, 0, stream>>>(xcat0, 1664, Wc0, 1664, bih0, bhh0,
                                                g0, 2048, 2048, 1664);
    lstm_act_kernel<<<BB, 512, 0, stream>>>(g0, c0, o_hN, o_cN, xcat1, 1024);
    copy_h01_kernel<<<BB, 512, 0, stream>>>(h0, xcat1);

    // 4-5. LSTM layer 1
    gemm64_kernel<false><<<8, 256, 0, stream>>>(xcat1, 1024, Wc1, 1024, bih1, bhh1,
                                                g1, 2048, 2048, 1024);
    lstm_act_kernel<<<BB, 512, 0, stream>>>(g1, c0 + BB * HH, o_hN + BB * HH,
                                            o_cN + BB * HH, out_in, 1664);

    // 6. dec_feat = s @ att_Wd^T  (s == out_in[:, 0:512])
    gemm64_kernel<false><<<2, 256, 0, stream>>>(out_in, 1664, Wdb, 512, nullptr, nullptr,
                                                decf, 512, 512, 512);

    // 7. fused attention energy/score GEMM (131072x1024 x 1024x512)
    attn_scores_kernel<<<dim3(2048, 2), 256, 0, stream>>>(enc, Whb, decf, cov,
                                                          attWc, attv, scores);

    // 8. attention softmax + coverage update
    attn_softmax_kernel<<<BB, 256, 0, stream>>>(scores, mask, cov, o_attn, o_cov);

    // 9. context = attn @ encoder_outputs
    context_kernel<<<dim3(BB, ENCD / 256), 256, 0, stream>>>(o_attn, enc, o_ctx, out_in);

    // 10. p_gen
    pgen_kernel<<<BB, 256, 0, stream>>>(out_in, pgW, pgb, o_pgen);

    // 11. logits = out_in @ out_W^T + out_b  (out_W converted f32->bf16 in-kernel)
    gemm64_kernel<true><<<(VV + 255) / 256, 256, 0, stream>>>(out_in, 1664, outW, 1664,
                                                              outb, nullptr, logits,
                                                              VV, VV, 1664);

    // 12. vocab softmax scaled by p_gen
    vocab_softmax_kernel<<<BB, 256, 0, stream>>>(logits, o_pgen, o_final);

    // 13. pointer-copy scatter add
    scatter_copy_kernel<<<dim3(BB, SS / 256), 256, 0, stream>>>(eids, o_attn, o_pgen, o_final);
}
